// KANLayer_45664092291554
// MI455X (gfx1250) — matrix operand order repaired and checked by reading, not compiled, not run on hardware
//
#include <hip/hip_runtime.h>
#include <hip/hip_bf16.h>

typedef __attribute__((ext_vector_type(16))) _Float16 v16h;
typedef __attribute__((ext_vector_type(8)))  float    v8f;

#define D_MODEL 64
#define IN_DIM  32
#define OUT_DIM 32
#define SIZE_   (IN_DIM*OUT_DIM)   // 1024
#define NB      8                  // spline bases per sample (NUM+K)
#define KW      9                  // per-i K slots: 8 bases + 1 silu
#define KDIM    (IN_DIM*KW)        // 288
#define KTILES  (KDIM/32)          // 9
#define BATCH   128
#define BPB     64                 // batches per block
#define NTILES  (BPB/16)           // 4

// Fused KAN layer: per (d, batch-half) workgroup.
//  Stage 1: build W (32x288) in LDS, pre-swizzled into WMMA A-fragment layout.
//  Stage 2: evaluate cubic B-spline bases + silu for 64 batches x 32 inputs,
//           pre-swizzled into WMMA B-fragment layout in LDS.
//  Stage 3: 2 mtiles x 4 ntiles of v_wmma_f32_16x16x32_f16 (K=288 -> 9 steps),
//           one mtile x 2 ntiles per wave (4 waves).
__global__ __launch_bounds__(128)
void kan_fused(const float* __restrict__ X,   // (B, D, 32)
               const float* __restrict__ G,   // (D, 1024, 6)
               const float* __restrict__ C,   // (D, 1024, 8)
               const float* __restrict__ Sb,  // (D, 1024)
               const float* __restrict__ Sp,  // (D, 1024)
               const float* __restrict__ Mk,  // (D, 1024)
               float* __restrict__ Out)       // (B, D, 32)
{
    __shared__ __attribute__((aligned(32))) _Float16 lW[KTILES][2][512];      // 18 KB
    __shared__ __attribute__((aligned(32))) _Float16 lB[KTILES][NTILES][512]; // 36 KB

    const int d     = blockIdx.y;
    const int bbase = blockIdx.x * BPB;
    const int tid   = threadIdx.x;
    const int lane  = tid & 31;
    const int wave  = tid >> 5;

    // ---------- Stage 1: weights -> A-fragment layout ----------
    // A 16x32 f16 layout: lanes 0-15 hold K 0..7 (j=0..7) and 16..23 (j=8..15);
    // lanes 16-31 hold K 8..15 and 24..31. Row M = lane%16.
    for (int e = tid; e < OUT_DIM * KDIM; e += 128) {
        int o = e / KDIM;
        int k = e - o * KDIM;
        int i = k / KW;
        int c = k - i * KW;
        int s = o * IN_DIM + i;
        float m = Mk[d * SIZE_ + s];
        float w = (c < NB) ? m * Sp[d * SIZE_ + s] * C[(d * SIZE_ + s) * NB + c]
                           : m * Sb[d * SIZE_ + s];
        int ktile  = k >> 5, krem = k & 31;
        int chunk  = krem >> 3, within = krem & 7;
        int wl     = (chunk & 1) * 16 + (o & 15);
        int wj     = (chunk >> 1) * 8 + within;
        lW[ktile][o >> 4][wl * 16 + wj] = (_Float16)w;
    }

    // ---------- Stage 2: bases + silu -> B-fragment layout ----------
    // B 32x16 f16 layout: column N = lane%16; lanes 0-15 hold K 0..7 (j=0..7) and 16..23 (j=8..15),
    // lanes 16-31 hold K 8..15 and 24..31 (same K order as A).
    for (int p = tid; p < IN_DIM * BPB; p += 128) {
        int i  = p & (IN_DIM - 1);
        int bl = p >> 5;               // 0..63 local batch
        int b  = bbase + bl;
        float x  = X[((size_t)b * D_MODEL + d) * IN_DIM + i];
        float g0 = G[((size_t)d * SIZE_ + i) * 6 + 0];
        float g5 = G[((size_t)d * SIZE_ + i) * 6 + 5];
        float h  = (g5 - g0) * 0.2f;   // uniform grid spacing
        float t[12];
        #pragma unroll
        for (int j = 0; j < 12; ++j) t[j] = g0 + (float)(j - 3) * h;
        float Bv[11];
        #pragma unroll
        for (int j = 0; j < 11; ++j)
            Bv[j] = (x >= t[j] && x < t[j + 1]) ? 1.0f : 0.0f;
        #pragma unroll
        for (int pd = 1; pd <= 3; ++pd) {
            float inv = 1.0f / ((float)pd * h);   // denominators are pd*h on uniform grid
            #pragma unroll
            for (int j = 0; j <= 10 - pd; ++j)
                Bv[j] = (x - t[j]) * inv * Bv[j] + (t[j + pd + 1] - x) * inv * Bv[j + 1];
        }
        float sil = x / (1.0f + __expf(-x));
        #pragma unroll
        for (int c = 0; c < KW; ++c) {
            float v = (c < NB) ? Bv[c] : sil;
            int k     = i * KW + c;
            int ktile = k >> 5, krem = k & 31;
            int wl    = ((krem >> 3) & 1) * 16 + (bl & 15);
            int wj    = (krem >> 4) * 8 + (krem & 7);
            lB[ktile][bl >> 4][wl * 16 + wj] = (_Float16)v;
        }
    }

    __syncthreads();

    // ---------- Stage 3: WMMA GEMM, K = 288 ----------
    const int mtile = wave & 1;
    const int ntb   = (wave >> 1) * 2;
    v8f acc0 = {};
    v8f acc1 = {};
    #pragma unroll
    for (int kt = 0; kt < KTILES; ++kt) {
        v16h a  = *(const v16h*)&lW[kt][mtile  ][lane * 16];
        v16h b0 = *(const v16h*)&lB[kt][ntb    ][lane * 16];
        v16h b1 = *(const v16h*)&lB[kt][ntb + 1][lane * 16];
        acc0 = __builtin_amdgcn_wmma_f32_16x16x32_f16(false, a, false, b0,
                                                      (short)0, acc0, false, false);
        acc1 = __builtin_amdgcn_wmma_f32_16x16x32_f16(false, a, false, b1,
                                                      (short)0, acc1, false, false);
    }

    // ---------- Epilogue: C/D layout -> (B, D, 32) ----------
    // VGPR r holds M = (lane<16 ? r : r+8), N = lane%16.
    int orow = mtile * 16 + (lane >> 4) * 8;
    int b0g  = bbase + ntb * 16 + (lane & 15);
    int b1g  = b0g + 16;
    float* p0 = Out + ((size_t)b0g * D_MODEL + d) * OUT_DIM + orow;
    float* p1 = Out + ((size_t)b1g * D_MODEL + d) * OUT_DIM + orow;
    #pragma unroll
    for (int r = 0; r < 8; ++r) {
        p0[r] = acc0[r];
        p1[r] = acc1[r];
    }
}

extern "C" void kernel_launch(void* const* d_in, const int* in_sizes, int n_in,
                              void* d_out, int out_size, void* d_ws, size_t ws_size,
                              hipStream_t stream) {
    (void)in_sizes; (void)n_in; (void)d_ws; (void)ws_size; (void)out_size;
    const float* X  = (const float*)d_in[0];
    const float* G  = (const float*)d_in[1];
    const float* C  = (const float*)d_in[2];
    const float* Sb = (const float*)d_in[3];
    const float* Sp = (const float*)d_in[4];
    const float* Mk = (const float*)d_in[5];
    // d_in[6] is k (==3), baked into the kernel.
    dim3 grid(BATCH / BPB, D_MODEL);   // (2, 64) = 128 workgroups
    kan_fused<<<grid, 128, 0, stream>>>(X, G, C, Sb, Sp, Mk, (float*)d_out);
}
